// DecoderRNN_84482006713137
// MI455X (gfx1250) — compile-verified, matrix operand order checked by reading, not run on hardware
//
#include <hip/hip_runtime.h>
#include <math.h>

#define HIDDEN   1024
#define EMBED    512
#define VOCAB    32000
#define SEQ      512
#define TSTEPS   511       // SEQ-1 recurrence steps
#define GATES    4096      // 4*HIDDEN
#define ROWS_PAD 512       // padded sequence rows for 16x16 tiling

#define RBLK 32            // persistent blocks in recurrence
#define RTH  128

// logits GEMM blocking: 64 rows x 256 cols per block, K staged via async->LDS
#define MROWS  64
#define BCOLS  256
#define NCHUNK 125         // 32000 / 256
#define KSTEP  32
#define KITER  (HIDDEN / KSTEP)

typedef _Float16 v8h  __attribute__((ext_vector_type(8)));
typedef _Float16 v16h __attribute__((ext_vector_type(16)));
typedef float    v8f  __attribute__((ext_vector_type(8)));

union Frag16 { v16h v; v8h h[2]; };

// 16-bit A/B fragment for V_WMMA_*_16X16X32_F16 (ISA 7.12.2):
// lane l (row/col = l&15) holds K = {off..off+7, 16+off..16+off+7}, off=(l>=16)*8
__device__ inline v16h load_frag(const _Float16* __restrict__ base, int ld,
                                 int mn, int kbase, int lane) {
  int r   = mn + (lane & 15);
  int off = (lane >> 4) * 8;
  const _Float16* p = base + (size_t)r * ld + kbase + off;
  Frag16 f;
  f.h[0] = *(const v8h*)(p);
  f.h[1] = *(const v8h*)(p + 16);
  return f.v;
}

__device__ inline v8f wmma_f16(v16h a, v16h b, v8f c) {
  return __builtin_amdgcn_wmma_f32_16x16x32_f16(false, a, false, b, (short)0, c,
                                                false, false);
}

// ---------------- weight conversion: fp32 -> f16 (halve L2 traffic) ----------
__global__ void k_f32_to_f16(const float* __restrict__ src,
                             _Float16* __restrict__ dst, int n) {
  int i = blockIdx.x * 256 + threadIdx.x;
  int stride = gridDim.x * 256;
  for (; i < n; i += stride) dst[i] = (_Float16)src[i];
}

// ---------------- init: h0 -> ping buffer, reset barrier counter -------------
__global__ void k_init(const float* __restrict__ h0, float* __restrict__ hbuf,
                       int* __restrict__ sync) {
  int i = blockIdx.x * blockDim.x + threadIdx.x;
  if (i < HIDDEN) hbuf[i] = h0[i];
  if (i == 0) sync[0] = 0;
}

// ---------------- embedding gather + relu, f16; zero pad rows ----------------
__global__ void k_embed(const float* __restrict__ emb, const int* __restrict__ y,
                        _Float16* __restrict__ X, _Float16* __restrict__ H) {
  int t = blockIdx.x;                      // 0..511
  if (t < TSTEPS) {
    int tok = y[t];
    const float* e = emb + (size_t)tok * EMBED;
    for (int c = threadIdx.x; c < EMBED; c += blockDim.x) {
      float v = e[c];
      X[t * EMBED + c] = (_Float16)(v > 0.f ? v : 0.f);
    }
  } else {                                 // pad row 511
    for (int c = threadIdx.x; c < EMBED; c += blockDim.x)
      X[t * EMBED + c] = (_Float16)0.f;
    for (int c = threadIdx.x; c < HIDDEN; c += blockDim.x)
      H[(size_t)TSTEPS * HIDDEN + c] = (_Float16)0.f;
  }
}

// ---------------- Gih = X @ W_ih^T + b_ih + b_hh  (512 x 4096, K=512) --------
__global__ void k_gemm_gih(const _Float16* __restrict__ X,
                           const _Float16* __restrict__ Wih,
                           const float* __restrict__ b_ih,
                           const float* __restrict__ b_hh,
                           float* __restrict__ Gih) {
  int wave = threadIdx.x >> 5;
  int lane = threadIdx.x & 31;
  int tile = blockIdx.x * 4 + wave;        // 8192 tiles total
  int tm = (tile & 31) * 16;               // 32 row tiles
  int tn = (tile >> 5) * 16;               // 256 col tiles
  v8f acc = {0.f, 0.f, 0.f, 0.f, 0.f, 0.f, 0.f, 0.f};
  for (int k = 0; k < EMBED; k += 32) {
    v16h a = load_frag(X,   EMBED, tm, k, lane);
    v16h b = load_frag(Wih, EMBED, tn, k, lane);
    acc = wmma_f16(a, b, acc);
  }
  int n = tn + (lane & 15);
  float bias = b_ih[n] + b_hh[n];
  int mhi = (lane >> 4) * 8;
#pragma unroll
  for (int r = 0; r < 8; ++r) {
    int m = tm + r + mhi;
    Gih[(size_t)m * GATES + n] = acc[r] + bias;
  }
}

// ---------------- serial LSTM recurrence, 32 persistent blocks ---------------
__device__ inline void grid_sync(int* cnt, int target) {
  __syncthreads();
  if (threadIdx.x == 0) {
    __threadfence();                       // release h writes
    atomicAdd(cnt, 1);
    while (__hip_atomic_load(cnt, __ATOMIC_ACQUIRE, __HIP_MEMORY_SCOPE_AGENT)
           < target) {
      __builtin_amdgcn_s_sleep(1);
    }
  }
  __syncthreads();
}

__global__ void k_recurrence(const _Float16* __restrict__ Whh,
                             const float* __restrict__ Gih,
                             float* __restrict__ hbuf,       // [2][HIDDEN] ping-pong
                             _Float16* __restrict__ H,       // [ROWS_PAD][HIDDEN]
                             int* __restrict__ sync) {
  int unit = blockIdx.x * 32 + (threadIdx.x >> 2);  // hidden unit 0..1023
  int gate = threadIdx.x & 3;                       // i,f,g,o
  int row  = gate * HIDDEN + unit;
  const _Float16* w = Whh + (size_t)row * HIDDEN;

  __shared__ float hs[HIDDEN];
  __shared__ float gv[32][4];
  float c = 0.f;

  for (int t = 0; t < TSTEPS; ++t) {
    const float* hr = hbuf + (size_t)(t & 1) * HIDDEN;
    float*       hw = hbuf + (size_t)((t + 1) & 1) * HIDDEN;
    for (int k = threadIdx.x; k < HIDDEN; k += RTH) hs[k] = hr[k];
    __syncthreads();

    float acc = Gih[(size_t)t * GATES + row];
    for (int k = 0; k < HIDDEN; k += 8) {
      v8h wv = *(const v8h*)(w + k);                // 16B f16 loads, L2-resident
#pragma unroll
      for (int j = 0; j < 8; ++j) acc += (float)wv[j] * hs[k + j];
    }
    gv[threadIdx.x >> 2][gate] = acc;
    __syncthreads();

    if (gate == 0) {
      int u = threadIdx.x >> 2;
      float gi = gv[u][0], gf = gv[u][1], gg = gv[u][2], go = gv[u][3];
      float si = 1.f / (1.f + __expf(-gi));
      float sf = 1.f / (1.f + __expf(-gf));
      float so = 1.f / (1.f + __expf(-go));
      c = sf * c + si * tanhf(gg);
      float h = so * tanhf(c);
      hw[unit] = h;
      H[(size_t)t * HIDDEN + unit] = (_Float16)h;
    }
    grid_sync(sync, RBLK * (t + 1));
  }
}

// -------- logits GEMM: 64x256 block tile, async->LDS double-buffered B ------
// fused online log-softmax partials (never materializes the 65MB logits)
__global__ void k_logits(const _Float16* __restrict__ H,
                         const _Float16* __restrict__ Wout,
                         const float* __restrict__ b_out,
                         const int* __restrict__ y,
                         float* __restrict__ pmax, float* __restrict__ psum,
                         float* __restrict__ pamaxv, int* __restrict__ pamaxi,
                         float* __restrict__ tlogit) {
  __shared__ _Float16 bbuf[2][BCOLS * KSTEP];   // 2 x 16 KB W_out tile
  __shared__ float lt[16][BCOLS];               // 16 KB epilogue staging
  __shared__ float smax[16][8], ssum[16][8];
  __shared__ int   sami[16][8];

  int rowBase = blockIdx.x * MROWS;   // 0..448
  int chunk   = blockIdx.y;           // 0..124
  int cb = chunk * BCOLS;
  int tid = threadIdx.x;
  int wave = tid >> 5, lane = tid & 31;

  const _Float16* wbase = Wout + (size_t)cb * HIDDEN;

  // Stage 256 rows x 32 halves of W_out for K-step ks into bbuf[b]:
  // GLOBAL_LOAD_ASYNC_TO_LDS_B128, tracked by ASYNCcnt (in-order completion).
  auto issue = [&](int b, int ks) {
    uint32_t ldsBase = (uint32_t)(uintptr_t)&bbuf[b][0];
    int kbase = ks * KSTEP;
#pragma unroll
    for (int i = 0; i < 8; ++i) {
      int idx = i * 128 + tid;       // 0..1023 : 16B segments
      int r   = idx >> 2;            // W_out row within chunk (0..255)
      int seg = idx & 3;
      uint64_t ga = (uint64_t)(uintptr_t)(wbase + (size_t)r * HIDDEN + kbase + seg * 8);
      uint32_t la = ldsBase + (uint32_t)(r * 64 + seg * 16);
      asm volatile("global_load_async_to_lds_b128 %0, %1, off"
                   :: "v"(la), "v"(ga) : "memory");
    }
  };

  v8f acc[4][4];
#pragma unroll
  for (int mt = 0; mt < 4; ++mt)
#pragma unroll
    for (int nt = 0; nt < 4; ++nt)
      acc[mt][nt] = (v8f){0.f,0.f,0.f,0.f,0.f,0.f,0.f,0.f};

  issue(0, 0);
  for (int ks = 0; ks < KITER; ++ks) {
    if (ks + 1 < KITER) {
      issue((ks + 1) & 1, ks + 1);
      asm volatile("s_wait_asynccnt 0x8" ::: "memory");  // prev batch of 8 done
    } else {
      asm volatile("s_wait_asynccnt 0x0" ::: "memory");
    }
    __syncthreads();

    const _Float16* lb = &bbuf[ks & 1][0];
    int kbase = ks * KSTEP;
    v16h aF[4];
#pragma unroll
    for (int mt = 0; mt < 4; ++mt)               // A frags reused over 4 N tiles
      aF[mt] = load_frag(H, HIDDEN, rowBase + mt * 16, kbase, lane);
#pragma unroll
    for (int nt = 0; nt < 4; ++nt) {
      int c   = (wave * 4 + nt) * 16 + (lane & 15);
      int off = (lane >> 4) * 8;
      const _Float16* p = lb + c * KSTEP + off;  // ds_load_b128 x2
      Frag16 f;
      f.h[0] = *(const v8h*)(p);
      f.h[1] = *(const v8h*)(p + 16);
#pragma unroll
      for (int mt = 0; mt < 4; ++mt)             // B frag reused over 4 M tiles
        acc[mt][nt] = wmma_f16(aF[mt], f.v, acc[mt][nt]);
    }
    __syncthreads();                             // protect buffer being refilled
  }

  // epilogue + fused partial log-softmax, 16 rows (one M tile) at a time
  int nl = lane & 15, mhi = (lane >> 4) * 8;
  for (int mt = 0; mt < 4; ++mt) {
#pragma unroll
    for (int nt = 0; nt < 4; ++nt) {
      int lc = (wave * 4 + nt) * 16 + nl;
      float bo = b_out[cb + lc];
#pragma unroll
      for (int r = 0; r < 8; ++r) {
        float v = acc[mt][nt][r] + bo;
        lt[r + mhi][lc] = v > 0.f ? v : 0.f;     // relu
      }
    }
    __syncthreads();

    int row = tid >> 3;            // 16 rows, 8 threads per row
    int sub = tid & 7;             // 32 cols per thread
    int rt  = rowBase + mt * 16 + row;
    float mx = -1e30f; int mi = cb + sub * 32;
    for (int c2 = sub * 32; c2 < sub * 32 + 32; ++c2) {
      float v = lt[row][c2];
      if (v > mx) { mx = v; mi = cb + c2; }
    }
    float se = 0.f;
    for (int c2 = sub * 32; c2 < sub * 32 + 32; ++c2)
      se += __expf(lt[row][c2] - mx);
    smax[row][sub] = mx; ssum[row][sub] = se; sami[row][sub] = mi;

    if (rt < TSTEPS) {                           // capture target logit
      int n = y[rt + 1];
      int loc = n - cb;
      if (loc >= sub * 32 && loc < sub * 32 + 32) tlogit[rt] = lt[row][loc];
    }
    __syncthreads();

    if (tid < 16 && (rowBase + mt * 16 + tid) < TSTEPS) {
      int rr = tid;
      float M = -1e30f;
      for (int s2 = 0; s2 < 8; ++s2) M = fmaxf(M, smax[rr][s2]);
      float S = 0.f, av = -1e30f; int ai = 0;
      for (int s2 = 0; s2 < 8; ++s2) {
        S += ssum[rr][s2] * __expf(smax[rr][s2] - M);
        if (smax[rr][s2] > av) { av = smax[rr][s2]; ai = sami[rr][s2]; }
      }
      int idx = (rowBase + mt * 16 + rr) * NCHUNK + chunk;
      pmax[idx] = M; psum[idx] = S; pamaxv[idx] = av; pamaxi[idx] = ai;
    }
    __syncthreads();
  }
}

// ---------------- merge chunk partials -> per-step loss / argmax -------------
__global__ void k_merge(const float* __restrict__ pmax, const float* __restrict__ psum,
                        const float* __restrict__ pamaxv, const int* __restrict__ pamaxi,
                        const float* __restrict__ tlogit,
                        float* __restrict__ loss, int* __restrict__ pred) {
  int rt = blockIdx.x * blockDim.x + threadIdx.x;
  if (rt >= TSTEPS) return;
  float M = -1e30f;
  for (int c = 0; c < NCHUNK; ++c) M = fmaxf(M, pmax[rt * NCHUNK + c]);
  float S = 0.f, av = -1e30f; int ai = 0;
  for (int c = 0; c < NCHUNK; ++c) {
    S += psum[rt * NCHUNK + c] * __expf(pmax[rt * NCHUNK + c] - M);
    float v = pamaxv[rt * NCHUNK + c];
    if (v > av) { av = v; ai = pamaxi[rt * NCHUNK + c]; }
  }
  loss[rt] = -(tlogit[rt] - M - __logf(S));
  pred[rt] = ai;
}

// ---------------- final: sum losses, emit [loss, SOS, pred...] ---------------
__global__ void k_final(const float* __restrict__ loss, const int* __restrict__ pred,
                        float* __restrict__ out, int out_size) {
  __shared__ float red[512];
  int i = threadIdx.x;
  red[i] = (i < TSTEPS) ? loss[i] : 0.f;
  __syncthreads();
  for (int s = 256; s > 0; s >>= 1) {
    if (i < s) red[i] += red[i + s];
    __syncthreads();
  }
  if (i == 0 && out_size > 0) out[0] = red[0];
  if (1 + i < out_size) out[1 + i] = (i == 0) ? 1.0f : (float)pred[i - 1];
}

// =============================== host side ===================================
static inline size_t alignup(size_t x) { return (x + 255) & ~(size_t)255; }

extern "C" void kernel_launch(void* const* d_in, const int* in_sizes, int n_in,
                              void* d_out, int out_size, void* d_ws, size_t ws_size,
                              hipStream_t stream) {
  const float* h0    = (const float*)d_in[0];   // (1024,)
  const float* emb   = (const float*)d_in[1];   // (32000, 512)
  const float* W_ih  = (const float*)d_in[2];   // (4096, 512)
  const float* W_hh  = (const float*)d_in[3];   // (4096, 1024)
  const float* b_ih  = (const float*)d_in[4];   // (4096,)
  const float* b_hh  = (const float*)d_in[5];   // (4096,)
  const float* W_out = (const float*)d_in[6];   // (32000, 1024)
  const float* b_out = (const float*)d_in[7];   // (32000,)
  const int*   y     = (const int*)d_in[8];     // (512,)
  float* out = (float*)d_out;

  // workspace carve-up
  char* p = (char*)d_ws; size_t off = 0;
  auto take = [&](size_t bytes) { char* r = p + off; off = alignup(off + bytes); return r; };
  _Float16* Wih_h  = (_Float16*)take((size_t)GATES * EMBED  * 2);
  _Float16* Whh_h  = (_Float16*)take((size_t)GATES * HIDDEN * 2);
  _Float16* Wout_h = (_Float16*)take((size_t)VOCAB * HIDDEN * 2);
  _Float16* X_h    = (_Float16*)take((size_t)ROWS_PAD * EMBED  * 2);
  float*    Gih    = (float*)take((size_t)ROWS_PAD * GATES * 4);
  _Float16* H_h    = (_Float16*)take((size_t)ROWS_PAD * HIDDEN * 2);
  float*    hbuf   = (float*)take(2 * HIDDEN * 4);
  int*      syncc  = (int*)take(256);
  float*    pmax   = (float*)take((size_t)ROWS_PAD * NCHUNK * 4);
  float*    psum   = (float*)take((size_t)ROWS_PAD * NCHUNK * 4);
  float*    pamaxv = (float*)take((size_t)ROWS_PAD * NCHUNK * 4);
  int*      pamaxi = (int*)take((size_t)ROWS_PAD * NCHUNK * 4);
  float*    tlog   = (float*)take(ROWS_PAD * 4);
  float*    loss   = (float*)take(ROWS_PAD * 4);
  int*      pred   = (int*)take(ROWS_PAD * 4);
  (void)ws_size; (void)n_in; (void)in_sizes;

  // 1) f16 weight copies (L2-resident working set: ~78 MB of 192 MB)
  k_f32_to_f16<<<4096, 256, 0, stream>>>(W_ih,  Wih_h,  GATES * EMBED);
  k_f32_to_f16<<<4096, 256, 0, stream>>>(W_hh,  Whh_h,  GATES * HIDDEN);
  k_f32_to_f16<<<8192, 256, 0, stream>>>(W_out, Wout_h, VOCAB * HIDDEN);

  // 2) init h0 / barrier counter
  k_init<<<4, 256, 0, stream>>>(h0, hbuf, syncc);

  // 3) embedding gather + relu
  k_embed<<<ROWS_PAD, 128, 0, stream>>>(emb, y, X_h, H_h);

  // 4) batched input-gate GEMM (WMMA f16)
  k_gemm_gih<<<2048, 128, 0, stream>>>(X_h, Wih_h, b_ih, b_hh, Gih);

  // 5) serial recurrence across 32 persistent workgroups
  k_recurrence<<<RBLK, RTH, 0, stream>>>(Whh_h, Gih, hbuf, H_h, syncc);

  // 6) logits GEMM + fused online log-softmax partials (WMMA + async->LDS)
  k_logits<<<dim3(ROWS_PAD / MROWS, NCHUNK), 128, 0, stream>>>(
      H_h, Wout_h, b_out, y, pmax, psum, pamaxv, pamaxi, tlog);

  // 7) merge partials -> loss_t, pred_t
  k_merge<<<2, 256, 0, stream>>>(pmax, psum, pamaxv, pamaxi, tlog, loss, pred);

  // 8) final reduction + output words
  k_final<<<1, 512, 0, stream>>>(loss, pred, out, out_size);
}
